// SimpleE3DiffusionModel_39651138077246
// MI455X (gfx1250) — matrix operand order repaired and checked by reading, not compiled, not run on hardware
//
#include <hip/hip_runtime.h>

#define N_NODES 10000
#define N_EDGES 320000
#define HID 128
#define N_TYPES 119
#define N_LAYERS 6

typedef __attribute__((ext_vector_type(16))) __bf16 v16bf;
typedef __attribute__((ext_vector_type(8)))  float  v8f;

#define WMMA_BF16(A, B, C) \
  __builtin_amdgcn_wmma_f32_16x16x32_bf16(false, (A), false, (B), (short)0, (C), false, false)

__device__ __forceinline__ unsigned short f2bfu(float f) {
  unsigned u = __builtin_bit_cast(unsigned, f);
  u += 0x7FFFu + ((u >> 16) & 1u);          // round-to-nearest-even
  return (unsigned short)(u >> 16);
}
__device__ __forceinline__ __bf16 bfu(unsigned short u) {
  return __builtin_bit_cast(__bf16, u);
}
// silu via v_exp_f32 + v_rcp_f32 (avoid IEEE divide expansion)
__device__ __forceinline__ float silu_f(float v) {
  return v * __builtin_amdgcn_rcpf(1.0f + __expf(-v));
}

// A-fragment (16xK bf16 tile) from an LDS tile, per CDNA5 16-bit A layout.
template<int LDC>
__device__ __forceinline__ v16bf load_a(const unsigned short (*s)[LDC], int lane, int kt) {
  const int m  = lane & 15;
  const int ko = kt * 32 + ((lane & 16) ? 8 : 0);
  v16bf a;
#pragma unroll
  for (int e = 0; e < 8; ++e) a[e] = bfu(s[m][ko + e]);
#pragma unroll
  for (int e = 0; e < 8; ++e) a[8 + e] = bfu(s[m][ko + 16 + e]);
  return a;
}

// B-fragment from pre-swizzled weights: layout [kt][ntile(8)][lane(32)][e(16)],
// each lane's 16 bf16 are 32B contiguous -> two coalesced global_load_b128.
__device__ __forceinline__ v16bf load_b_sw(const unsigned short* __restrict__ W,
                                           int wv, int lane, int kt) {
  const unsigned short* p = W + (size_t)((((kt << 3) + wv) << 5) + lane) * 16;
  v16bf b;
#pragma unroll
  for (int e = 0; e < 16; ++e) b[e] = bfu(p[e]);
  return b;
}

// Async 16B global->LDS copy (CDNA5, ASYNCcnt-tracked). Generic LDS pointers
// hold the LDS byte offset in their low 32 bits (flat aperture rule).
__device__ __forceinline__ void async_copy16(const void* gsrc, void* ldst) {
  unsigned lds_off = (unsigned)(unsigned long long)ldst;
  asm volatile("global_load_async_to_lds_b128 %0, %1, off"
               :: "v"(lds_off), "v"(gsrc) : "memory");
}
__device__ __forceinline__ void async_wait0() {
  asm volatile("s_wait_asynccnt 0x0" ::: "memory");
}

// ---------------- small helper kernels ----------------

__global__ void k_zero(float* __restrict__ p, int n) {
  for (int i = blockIdx.x * blockDim.x + threadIdx.x; i < n; i += gridDim.x * blockDim.x)
    p[i] = 0.0f;
}

// fp32 -> bf16 weight conversion into fragment-swizzled layout.
// src: [layers][rows_src][128]; dst: [layers][kt_count][8][32][16] bf16,
// element (kt,ntile,lane,e) = W[kt*32 + (lane&16?8:0) + (e&7) + (e&8?16:0)][ntile*16 + (lane&15)]
// (zero-padded for krow >= rows_src).
__global__ void k_cvt_sw(const float* __restrict__ src, unsigned short* __restrict__ dst,
                         int rows_src, int kt_count, int total) {
  const int per_layer = kt_count << 12;                 // kt_count * 4096
  for (int i = blockIdx.x * blockDim.x + threadIdx.x; i < total; i += gridDim.x * blockDim.x) {
    const int l = i / per_layer;
    const int rem = i % per_layer;
    const int kt = rem >> 12;
    const int r2 = rem & 4095;
    const int ntile = r2 >> 9;
    const int lane = (r2 >> 4) & 31;
    const int e = r2 & 15;
    const int krow = kt * 32 + ((lane & 16) ? 8 : 0) + (e & 7) + ((e & 8) ? 16 : 0);
    const int col = (ntile << 4) | (lane & 15);
    const float v = (krow < rows_src) ? src[((size_t)l * rows_src + krow) * HID + col] : 0.0f;
    dst[i] = f2bfu(v);
  }
}

__global__ __launch_bounds__(128) void k_time(const float* __restrict__ t,
    const float* __restrict__ tw1, const float* __restrict__ tb1,
    const float* __restrict__ tw2, const float* __restrict__ tb2,
    float* __restrict__ te) {
  __shared__ float su[128];
  const int tid = threadIdx.x;
  su[tid] = silu_f(t[0] * tw1[tid] + tb1[tid]);
  __syncthreads();
  float s = tb2[tid];
#pragma unroll 8
  for (int k = 0; k < 128; ++k) s += su[k] * tw2[k * 128 + tid];
  te[tid] = s;
}

__global__ void k_init(const int* __restrict__ x, const float* __restrict__ emb,
                       const float* __restrict__ te, float* __restrict__ h,
                       unsigned short* __restrict__ hb) {
  const int i = blockIdx.x * blockDim.x + threadIdx.x;
  if (i >= N_NODES * HID) return;
  const int node = i >> 7, c = i & 127;
  const float v = emb[x[node] * HID + c] + te[c];
  h[i] = v;
  hb[i] = f2bfu(v);
}

__global__ void k_count(const int* __restrict__ ei, float* __restrict__ deg) {
  const int e = blockIdx.x * blockDim.x + threadIdx.x;
  if (e < N_EDGES) atomicAdd(&deg[ei[N_EDGES + e]], 1.0f);
}

__global__ void k_invdeg(float* __restrict__ d) {
  const int i = blockIdx.x * blockDim.x + threadIdx.x;
  if (i < N_NODES) {
    const float v = d[i];
    d[i] = (v > 0.0f) ? 1.0f / v : 0.0f;
  }
}

// ---------------- edge-message MLP (WMMA) ----------------
// grid = E/128 blocks, 256 threads (8 waves). Each block: 8 tiles of 16 edges.
// Wave w owns output columns [16w,16w+16). W1 (288x128) + W2 (128x128)
// fragments live in VGPRs for the whole block. h rows staged to LDS via
// async global->LDS copies; GEMMs use dual accumulators + pipelined A loads.
__global__ __launch_bounds__(256) void k_edge(
    const unsigned short* __restrict__ hb, const float* __restrict__ pos,
    const int* __restrict__ ei,
    const unsigned short* __restrict__ w1, const float* __restrict__ b1,
    const unsigned short* __restrict__ w2, const float* __restrict__ b2,
    float* __restrict__ agg) {
  __shared__ unsigned short sA[16][288];
  __shared__ unsigned short sM[16][128];
  __shared__ int sDst[16];
  __shared__ int sSrc[16];

  const int tid = threadIdx.x;
  const int lane = tid & 31;
  const int wv = tid >> 5;
  const int colg = (wv << 4) | (lane & 15);
  const int mb = (lane & 16) ? 8 : 0;
  const int row = tid >> 4;     // staging: row 0..15
  const int chunk = tid & 15;   // staging: 16B chunk 0..15 within 128 cols

  v16bf B1[9];
#pragma unroll
  for (int kt = 0; kt < 9; ++kt) B1[kt] = load_b_sw(w1, wv, lane, kt);
  v16bf B2[4];
#pragma unroll
  for (int kt = 0; kt < 4; ++kt) B2[kt] = load_b_sw(w2, wv, lane, kt);
  const float bias1 = b1[colg];
  const float bias2 = b2[colg];
  const v8f vzero = {0, 0, 0, 0, 0, 0, 0, 0};

  for (int tile = 0; tile < 8; ++tile) {
    const int e0 = (blockIdx.x * 8 + tile) << 4;
    if (tid < 16) {
      int e = e0 + tid;
      if (e >= N_EDGES) e = N_EDGES - 1;
      sSrc[tid] = ei[e];
      sDst[tid] = ei[N_EDGES + e];
    }
    __syncthreads();
    // stage A tile: [h_dst(128) | h_src(128) | rel_pos(3) | pad 0 (29)]
    async_copy16(hb + (size_t)sDst[row] * HID + chunk * 8, &sA[row][chunk * 8]);
    async_copy16(hb + (size_t)sSrc[row] * HID + chunk * 8, &sA[row][HID + chunk * 8]);
    {
      // rel_pos + zero padding: cols 256..287, 16x32 elements, 2 per thread
      for (int i = tid; i < 16 * 32; i += 256) {
        const int r = i >> 5, c = 256 + (i & 31);
        unsigned short v = 0;
        if (c < 2 * HID + 3) {
          const int d = c - 2 * HID;
          v = f2bfu(pos[sSrc[r] * 3 + d] - pos[sDst[r] * 3 + d]);
        }
        sA[r][c] = v;
      }
    }
    async_wait0();
    __syncthreads();

    // GEMM1: 9 k-tiles, dual accumulators, A loads pipelined 2 ahead
    v8f acc0 = vzero, acc1 = vzero;
    {
      v16bf aA = load_a<288>(sA, lane, 0);
      v16bf aB = load_a<288>(sA, lane, 1);
#pragma unroll
      for (int kt = 0; kt < 9; kt += 2) {
        v16bf aN0 = (kt + 2 < 9) ? load_a<288>(sA, lane, kt + 2) : aA;
        v16bf aN1 = (kt + 3 < 9) ? load_a<288>(sA, lane, kt + 3) : aB;
        acc0 = WMMA_BF16(aA, B1[kt], acc0);
        if (kt + 1 < 9) acc1 = WMMA_BF16(aB, B1[kt + 1], acc1);
        aA = aN0; aB = aN1;
      }
    }
#pragma unroll
    for (int r = 0; r < 8; ++r)
      sM[mb + r][colg] = f2bfu(silu_f(acc0[r] + acc1[r] + bias1));
    __syncthreads();

    // GEMM2: 4 k-tiles, dual accumulators
    v8f c0 = vzero, c1 = vzero;
    {
      v16bf a0 = load_a<128>(sM, lane, 0);
      v16bf a1 = load_a<128>(sM, lane, 1);
      v16bf a2 = load_a<128>(sM, lane, 2);
      v16bf a3 = load_a<128>(sM, lane, 3);
      c0 = WMMA_BF16(a0, B2[0], c0);
      c1 = WMMA_BF16(a1, B2[1], c1);
      c0 = WMMA_BF16(a2, B2[2], c0);
      c1 = WMMA_BF16(a3, B2[3], c1);
    }
#pragma unroll
    for (int r = 0; r < 8; ++r)
      atomicAdd(&agg[sDst[mb + r] * HID + colg], c0[r] + c1[r] + bias2);
    __syncthreads();
  }
}

// ---------------- node-update MLP (WMMA) ----------------
// grid = N/16 blocks, 256 threads. A = [agg*inv_deg(128) | h(128)].
__global__ __launch_bounds__(256) void k_update(
    unsigned short* __restrict__ hb, float* __restrict__ h,
    const float* __restrict__ agg, const float* __restrict__ invdeg,
    const unsigned short* __restrict__ w1, const float* __restrict__ b1,
    const unsigned short* __restrict__ w2, const float* __restrict__ b2) {
  __shared__ unsigned short sA[16][256];
  __shared__ unsigned short sU[16][128];

  const int tid = threadIdx.x;
  const int lane = tid & 31;
  const int wv = tid >> 5;
  const int colg = (wv << 4) | (lane & 15);
  const int mb = (lane & 16) ? 8 : 0;
  const int n0 = blockIdx.x << 4;

  v16bf B1[8];
#pragma unroll
  for (int kt = 0; kt < 8; ++kt) B1[kt] = load_b_sw(w1, wv, lane, kt);
  v16bf B2[4];
#pragma unroll
  for (int kt = 0; kt < 4; ++kt) B2[kt] = load_b_sw(w2, wv, lane, kt);
  const float bias1 = b1[colg];
  const float bias2 = b2[colg];
  const v8f vzero = {0, 0, 0, 0, 0, 0, 0, 0};

  for (int i = tid; i < 16 * 256; i += 256) {
    const int r = i >> 8, c = i & 255;
    const int node = n0 + r;
    unsigned short v;
    if (c < HID) v = f2bfu(agg[node * HID + c] * invdeg[node]);
    else         v = hb[node * HID + (c - HID)];
    sA[r][c] = v;
  }
  __syncthreads();

  v8f acc0 = vzero, acc1 = vzero;
  {
    v16bf aA = load_a<256>(sA, lane, 0);
    v16bf aB = load_a<256>(sA, lane, 1);
#pragma unroll
    for (int kt = 0; kt < 8; kt += 2) {
      v16bf aN0 = (kt + 2 < 8) ? load_a<256>(sA, lane, kt + 2) : aA;
      v16bf aN1 = (kt + 3 < 8) ? load_a<256>(sA, lane, kt + 3) : aB;
      acc0 = WMMA_BF16(aA, B1[kt], acc0);
      acc1 = WMMA_BF16(aB, B1[kt + 1], acc1);
      aA = aN0; aB = aN1;
    }
  }
#pragma unroll
  for (int r = 0; r < 8; ++r)
    sU[mb + r][colg] = f2bfu(silu_f(acc0[r] + acc1[r] + bias1));
  __syncthreads();

  v8f c0 = vzero, c1 = vzero;
  {
    v16bf a0 = load_a<128>(sU, lane, 0);
    v16bf a1 = load_a<128>(sU, lane, 1);
    v16bf a2 = load_a<128>(sU, lane, 2);
    v16bf a3 = load_a<128>(sU, lane, 3);
    c0 = WMMA_BF16(a0, B2[0], c0);
    c1 = WMMA_BF16(a1, B2[1], c1);
    c0 = WMMA_BF16(a2, B2[2], c0);
    c1 = WMMA_BF16(a3, B2[3], c1);
  }
#pragma unroll
  for (int r = 0; r < 8; ++r) {
    const int node = n0 + mb + r;
    const float v = c0[r] + c1[r] + bias2;
    h[node * HID + colg] = v;
    hb[node * HID + colg] = f2bfu(v);
  }
}

// ---------------- output heads ----------------
__global__ __launch_bounds__(128) void k_head(
    const float* __restrict__ h,
    const float* __restrict__ aw, const float* __restrict__ ab,
    const float* __restrict__ pw, const float* __restrict__ pb,
    float* __restrict__ out) {
  __shared__ float sh[HID];
  const int node = blockIdx.x, tid = threadIdx.x;
  sh[tid] = h[node * HID + tid];
  __syncthreads();
  if (tid < N_TYPES) {
    float s = ab[tid];
#pragma unroll 8
    for (int k = 0; k < HID; ++k) s += sh[k] * aw[k * N_TYPES + tid];
    out[node * N_TYPES + tid] = s;
  } else if (tid < N_TYPES + 3) {
    const int c = tid - N_TYPES;
    float s = pb[c];
#pragma unroll 8
    for (int k = 0; k < HID; ++k) s += sh[k] * pw[k * 3 + c];
    out[(size_t)N_NODES * N_TYPES + node * 3 + c] = s;
  }
}

extern "C" void kernel_launch(void* const* d_in, const int* in_sizes, int n_in,
                              void* d_out, int out_size, void* d_ws, size_t ws_size,
                              hipStream_t stream) {
  (void)in_sizes; (void)n_in; (void)out_size; (void)ws_size;
  const int*   x   = (const int*)  d_in[0];
  const float* pos = (const float*)d_in[1];
  const int*   ei  = (const int*)  d_in[2];
  const float* t   = (const float*)d_in[3];
  const float* tw1 = (const float*)d_in[4];
  const float* tb1 = (const float*)d_in[5];
  const float* tw2 = (const float*)d_in[6];
  const float* tb2 = (const float*)d_in[7];
  const float* emb = (const float*)d_in[8];
  const float* mw1 = (const float*)d_in[9];
  const float* mb1 = (const float*)d_in[10];
  const float* mw2 = (const float*)d_in[11];
  const float* mb2 = (const float*)d_in[12];
  const float* uw1 = (const float*)d_in[13];
  const float* ub1 = (const float*)d_in[14];
  const float* uw2 = (const float*)d_in[15];
  const float* ub2 = (const float*)d_in[16];
  const float* haw = (const float*)d_in[17];
  const float* hab = (const float*)d_in[18];
  const float* hpw = (const float*)d_in[19];
  const float* hpb = (const float*)d_in[20];
  float* out = (float*)d_out;

  char* ws = (char*)d_ws;
  size_t off = 0;
  auto take = [&](size_t bytes) {
    char* p = ws + off;
    off = (off + bytes + 255) & ~(size_t)255;
    return p;
  };
  float* te            = (float*)take(128 * 4);
  float* h             = (float*)take((size_t)N_NODES * HID * 4);
  unsigned short* hb   = (unsigned short*)take((size_t)N_NODES * HID * 2);
  float* agg           = (float*)take((size_t)N_NODES * HID * 4);
  float* ideg          = (float*)take(N_NODES * 4);
  unsigned short* w1b  = (unsigned short*)take((size_t)N_LAYERS * 9 * 4096 * 2);
  unsigned short* w2b  = (unsigned short*)take((size_t)N_LAYERS * 4 * 4096 * 2);
  unsigned short* u1b  = (unsigned short*)take((size_t)N_LAYERS * 8 * 4096 * 2);
  unsigned short* u2b  = (unsigned short*)take((size_t)N_LAYERS * 4 * 4096 * 2);

  int tot = N_LAYERS * 9 * 4096;
  k_cvt_sw<<<(tot + 255) / 256, 256, 0, stream>>>(mw1, w1b, 2 * HID + 3, 9, tot);
  tot = N_LAYERS * 4 * 4096;
  k_cvt_sw<<<(tot + 255) / 256, 256, 0, stream>>>(mw2, w2b, HID, 4, tot);
  tot = N_LAYERS * 8 * 4096;
  k_cvt_sw<<<(tot + 255) / 256, 256, 0, stream>>>(uw1, u1b, 2 * HID, 8, tot);
  tot = N_LAYERS * 4 * 4096;
  k_cvt_sw<<<(tot + 255) / 256, 256, 0, stream>>>(uw2, u2b, HID, 4, tot);

  k_time<<<1, 128, 0, stream>>>(t, tw1, tb1, tw2, tb2, te);
  k_init<<<(N_NODES * HID) / 256, 256, 0, stream>>>(x, emb, te, h, hb);
  k_zero<<<40, 256, 0, stream>>>(ideg, N_NODES);
  k_count<<<N_EDGES / 256, 256, 0, stream>>>(ei, ideg);
  k_invdeg<<<40, 256, 0, stream>>>(ideg);

  for (int l = 0; l < N_LAYERS; ++l) {
    k_zero<<<(N_NODES * HID) / 256, 256, 0, stream>>>(agg, N_NODES * HID);
    k_edge<<<N_EDGES / 128, 256, 0, stream>>>(hb, pos, ei,
        w1b + (size_t)l * 9 * 4096, mb1 + l * HID,
        w2b + (size_t)l * 4 * 4096, mb2 + l * HID, agg);
    k_update<<<N_NODES / 16, 256, 0, stream>>>(hb, h, agg, ideg,
        u1b + (size_t)l * 8 * 4096, ub1 + l * HID,
        u2b + (size_t)l * 4 * 4096, ub2 + l * HID);
  }
  k_head<<<N_NODES, 128, 0, stream>>>(h, haw, hab, hpw, hpb, out);
}
